// ContextualLSTMTransformerFlexible_56229711839322
// MI455X (gfx1250) — matrix operand order repaired and checked by reading, not compiled, not run on hardware
//
#include <hip/hip_runtime.h>
#include <cstdint>
#include <cstddef>

// ---------- WMMA fragment helpers (gfx1250, wave32) ----------
typedef __attribute__((ext_vector_type(16))) _Float16 v16h;
typedef __attribute__((ext_vector_type(8)))  float    v8f;

struct __attribute__((aligned(16))) U128 { unsigned int a, b, c, d; };
union HV16 { v16h v; U128 q[2]; };

__device__ __forceinline__ int laneid() { return (int)(threadIdx.x & 31u); }

// A-matrix 16x32 f16 (ISA 7.12.2): lane m = lane&15; halfs 0-7 -> K = kbase+g0..+7,
// halfs 8-15 -> K = kbase+16+g0.. with g0 = (lane>=16 ? 8 : 0). Two 16B contiguous loads.
__device__ __forceinline__ v16h load_a_frag(const _Float16* base, int stride, int kbase) {
  int l = laneid();
  const _Float16* p = base + (l & 15) * stride + kbase + ((l >> 4) << 3);
  HV16 h;
  h.q[0] = *(const U128*)(p);
  h.q[1] = *(const U128*)(p + 16);
  return h.v;
}

// B-matrix 32x16 f16: lane column n = n0 + (lane&15); halfs 0-7 -> K = kbase+g0..+7,
// halfs 8-15 -> K = kbase+16+g0.. with g0 = (lane>=16 ? 8 : 0). Row-major W (N x K).
__device__ __forceinline__ v16h load_b_frag(const _Float16* w, int stride, int n0, int kbase) {
  int l = laneid();
  const _Float16* p = w + (size_t)(n0 + (l & 15)) * stride + kbase + ((l >> 4) << 3);
  HV16 h;
  h.q[0] = *(const U128*)(p);
  h.q[1] = *(const U128*)(p + 16);
  return h.v;
}

__device__ __forceinline__ v8f wmma32(v16h a, v16h b, v8f c) {
  return __builtin_amdgcn_wmma_f32_16x16x32_f16(false, a, false, b, (short)0, c, false, false);
}

__device__ __forceinline__ v8f splat8(float v) { v8f r = {v, v, v, v, v, v, v, v}; return r; }

// D fragment: VGPR r, lane l -> (m = m0 + r + (l>=16?8:0), n = n0 + (l&15))
__device__ __forceinline__ void store_d_f16(_Float16* dst, int stride, int m0, int n0, v8f c) {
  int l = laneid();
  int n = n0 + (l & 15);
  int mb = m0 + ((l >> 4) << 3);
#pragma unroll
  for (int r = 0; r < 8; ++r) dst[(mb + r) * stride + n] = (_Float16)c[r];
}

// hardware-rate rcp (single v_rcp_f32, no Newton refinement / div_fixup)
__device__ __forceinline__ float rcp_fast(float x) { return __builtin_amdgcn_rcpf(x); }

__device__ __forceinline__ float sigf(float x) {
  return rcp_fast(1.f + __expf(-x));
}

// branchless tanh: v_exp + v_rcp + bitops, no libm range-check divergence
__device__ __forceinline__ float tanh_fast(float x) {
  float ax = __builtin_fabsf(x);
  float e  = __expf(-2.f * ax);
  float r  = (1.f - e) * rcp_fast(1.f + e);
  return __builtin_copysignf(r, x);
}

// ---------- Model constants ----------
#define NWIN_TOT 16384   // B(256) * NW(64)
#define WIN_T    32
#define EDIM     128
#define HDIM     64

// ---------- K0: weight prep (f32 -> f16, pad w_ih K 16->32) ----------
__global__ __launch_bounds__(256) void k_prep(
    const float* __restrict__ wihf, const float* __restrict__ whhf,
    const float* __restrict__ bihf, const float* __restrict__ bhhf,
    const float* __restrict__ wihb, const float* __restrict__ whhb,
    const float* __restrict__ bihb, const float* __restrict__ bhhb,
    const float* __restrict__ sain, const float* __restrict__ saout,
    const float* __restrict__ projw,
    _Float16* __restrict__ wih16, _Float16* __restrict__ whh16,
    float* __restrict__ bsum, _Float16* __restrict__ sain16,
    _Float16* __restrict__ saout16, _Float16* __restrict__ proj16)
{
  int i = blockIdx.x * 256 + threadIdx.x;
  if (i < 16384) {                       // wih16[2][256][32], K padded
    int d = i >> 13, r = i & 8191;
    int g = r >> 5, k2 = r & 31;
    const float* src = d ? wihb : wihf;
    wih16[i] = (_Float16)(k2 < 16 ? src[g * 16 + k2] : 0.f);
    return;
  }
  i -= 16384;
  if (i < 32768) {                       // whh16[2][256][64]
    int d = i >> 14, r = i & 16383;
    const float* src = d ? whhb : whhf;
    whh16[i] = (_Float16)src[r];
    return;
  }
  i -= 32768;
  if (i < 512) {                         // bsum[2][256] = b_ih + b_hh
    int d = i >> 8, g = i & 255;
    bsum[i] = d ? (bihb[g] + bhhb[g]) : (bihf[g] + bhhf[g]);
    return;
  }
  i -= 512;
  if (i < 49152) { sain16[i]  = (_Float16)sain[i];  return; }
  i -= 49152;
  if (i < 16384) { saout16[i] = (_Float16)saout[i]; return; }
  i -= 16384;
  if (i < 16384) { proj16[i]  = (_Float16)projw[i]; return; }
}

// ---------- K1: fused BiLSTM, 16 windows per wave, WMMA recurrence ----------
__global__ __launch_bounds__(128) void k_bilstm(
    const float* __restrict__ x, const _Float16* __restrict__ wih,
    const _Float16* __restrict__ whh, const float* __restrict__ bsum,
    _Float16* __restrict__ lo)
{
  __shared__ _Float16 xs[4][16 * 32];
  __shared__ _Float16 hs[4][16 * 64];
  __shared__ float    cs[4][16 * 64];
  int wv = threadIdx.x >> 5, l = threadIdx.x & 31;
  int w0 = (blockIdx.x * 4 + wv) * 16;     // first window of this wave
  _Float16* xb = xs[wv];
  _Float16* hb = hs[wv];
  float*    cb = cs[wv];
  for (int i = l; i < 16 * 32; i += 32) xb[i] = (_Float16)0.f;  // feat pad stays 0
  int nn15 = l & 15;
  int mb = (l >> 4) << 3;

  for (int dir = 0; dir < 2; ++dir) {
    for (int i = l; i < 16 * 64; i += 32) { hb[i] = (_Float16)0.f; cb[i] = 0.f; }
    const _Float16* WI = wih + dir * 256 * 32;
    const _Float16* WH = whh + dir * 256 * 64;
    const float*    BS = bsum + dir * 256;
    for (int s = 0; s < 32; ++s) {
      int t = dir ? (31 - s) : s;          // bwd: reversed scan, output re-reversed
      for (int i = l; i < 256; i += 32) {  // stage x_t for 16 windows
        int m = i >> 4, f = i & 15;
        xb[m * 32 + f] = (_Float16)x[(size_t)(w0 + m) * 512 + t * 16 + f];
      }
      v16h ax  = load_a_frag(xb, 32, 0);
      v16h ah0 = load_a_frag(hb, 64, 0);
      v16h ah1 = load_a_frag(hb, 64, 32);
#pragma unroll
      for (int jj = 0; jj < 4; ++jj) {     // 16-col groups of H
        v8f g4[4];
#pragma unroll
        for (int g = 0; g < 4; ++g) {      // i, f, g, o gates
          int n0 = g * 64 + jj * 16;
          v8f c = splat8(BS[n0 + nn15]);
          c = wmma32(ax,  load_b_frag(WI, 32, n0, 0),  c);
          c = wmma32(ah0, load_b_frag(WH, 64, n0, 0),  c);
          c = wmma32(ah1, load_b_frag(WH, 64, n0, 32), c);
          g4[g] = c;
        }
        int hidx = jj * 16 + nn15;
#pragma unroll
        for (int r = 0; r < 8; ++r) {
          int m = mb + r;
          float iv = sigf((float)g4[0][r]);
          float fv = sigf((float)g4[1][r]);
          float gv = tanh_fast((float)g4[2][r]);
          float ov = sigf((float)g4[3][r]);
          float cv = fv * cb[m * 64 + hidx] + iv * gv;
          cb[m * 64 + hidx] = cv;
          float hv = ov * tanh_fast(cv);
          hb[m * 64 + hidx] = (_Float16)hv;
          lo[((size_t)(w0 + m) * 32 + t) * 128 + dir * 64 + hidx] = (_Float16)hv;
        }
      }
    }
  }
}

// ---------- K2: self-MHA + out proj + proj + mean-pool, one wave per window ----------
__global__ __launch_bounds__(32) void k_mha(
    const _Float16* __restrict__ lo,
    const _Float16* __restrict__ sain, const float* __restrict__ sainb,
    const _Float16* __restrict__ saout, const float* __restrict__ saoutb,
    const _Float16* __restrict__ projw, const float* __restrict__ projb,
    float* __restrict__ pooled)
{
  __shared__ _Float16 qb[32 * 128];
  __shared__ _Float16 kb[32 * 128];
  __shared__ _Float16 vT[128 * 32];   // v transposed: [channel][time]
  __shared__ _Float16 ob[32 * 128];
  __shared__ _Float16 sab[32 * 128];
  __shared__ float    sc[32 * 32];
  __shared__ _Float16 ab[32 * 32];
  int l = laneid();
  int w = blockIdx.x;
  const _Float16* loW = lo + (size_t)w * 32 * 128;
  int n15 = l & 15, mb = (l >> 4) << 3;

  // --- q, k, v projections: lo(32x128) @ W^T ---
  for (int mt = 0; mt < 2; ++mt) {
    v16h a[4];
#pragma unroll
    for (int kc = 0; kc < 4; ++kc) a[kc] = load_a_frag(loW + mt * 16 * 128, 128, kc * 32);
    for (int part = 0; part < 3; ++part) {
      const _Float16* W = sain + part * 128 * 128;
      const float* bb = sainb + part * 128;
      for (int nt = 0; nt < 8; ++nt) {
        v8f c = splat8(bb[nt * 16 + n15]);
#pragma unroll
        for (int kc = 0; kc < 4; ++kc) c = wmma32(a[kc], load_b_frag(W, 128, nt * 16, kc * 32), c);
        if (part == 0)      store_d_f16(qb, 128, mt * 16, nt * 16, c);
        else if (part == 1) store_d_f16(kb, 128, mt * 16, nt * 16, c);
        else {
#pragma unroll
          for (int r = 0; r < 8; ++r)
            vT[(nt * 16 + n15) * 32 + (mt * 16 + mb + r)] = (_Float16)c[r];
        }
      }
    }
  }

  // --- attention per head (d = 32) ---
  for (int h = 0; h < 4; ++h) {
    for (int mt = 0; mt < 2; ++mt) {
      v16h aq = load_a_frag(qb + mt * 16 * 128, 128, h * 32);
      for (int nt = 0; nt < 2; ++nt) {
        v8f c = splat8(0.f);
        c = wmma32(aq, load_b_frag(kb, 128, nt * 16, h * 32), c);
#pragma unroll
        for (int r = 0; r < 8; ++r)
          sc[(mt * 16 + mb + r) * 32 + nt * 16 + n15] = (float)c[r] * 0.17677669529663687f;
      }
    }
    // softmax: lane l owns row l (32 rows, 32 lanes)
    float mx = -3.0e38f;
    for (int c2 = 0; c2 < 32; ++c2) mx = fmaxf(mx, sc[l * 32 + c2]);
    float ssum = 0.f;
    for (int c2 = 0; c2 < 32; ++c2) { float e = __expf(sc[l * 32 + c2] - mx); ssum += e; sc[l * 32 + c2] = e; }
    float inv = rcp_fast(ssum);
    for (int c2 = 0; c2 < 32; ++c2) ab[l * 32 + c2] = (_Float16)(sc[l * 32 + c2] * inv);
    // o = a @ v  (B from transposed v: contiguous k per lane)
    for (int mt = 0; mt < 2; ++mt) {
      v16h aa = load_a_frag(ab + mt * 16 * 32, 32, 0);
      for (int nt = 0; nt < 2; ++nt) {
        int n0 = h * 32 + nt * 16;
        v8f c = splat8(0.f);
        c = wmma32(aa, load_b_frag(vT, 32, n0, 0), c);
        store_d_f16(ob, 128, mt * 16, n0, c);
      }
    }
  }

  // --- sa = ob @ sa_out_w^T + b ---
  for (int mt = 0; mt < 2; ++mt) {
    v16h a[4];
#pragma unroll
    for (int kc = 0; kc < 4; ++kc) a[kc] = load_a_frag(ob + mt * 16 * 128, 128, kc * 32);
    for (int nt = 0; nt < 8; ++nt) {
      v8f c = splat8(saoutb[nt * 16 + n15]);
#pragma unroll
      for (int kc = 0; kc < 4; ++kc) c = wmma32(a[kc], load_b_frag(saout, 128, nt * 16, kc * 32), c);
      store_d_f16(sab, 128, mt * 16, nt * 16, c);
    }
  }

  // --- p = sab @ proj_w^T + proj_b ; mean over 32 rows -> pooled[w][:] ---
  float ps[8];
#pragma unroll
  for (int nt = 0; nt < 8; ++nt) ps[nt] = 0.f;
  for (int mt = 0; mt < 2; ++mt) {
    v16h a[4];
#pragma unroll
    for (int kc = 0; kc < 4; ++kc) a[kc] = load_a_frag(sab + mt * 16 * 128, 128, kc * 32);
#pragma unroll
    for (int nt = 0; nt < 8; ++nt) {
      v8f c = splat8(0.f);
#pragma unroll
      for (int kc = 0; kc < 4; ++kc) c = wmma32(a[kc], load_b_frag(projw, 128, nt * 16, kc * 32), c);
#pragma unroll
      for (int r = 0; r < 8; ++r) ps[nt] += (float)c[r];
    }
  }
#pragma unroll
  for (int nt = 0; nt < 8; ++nt) {
    float v = ps[nt] + __shfl_xor(ps[nt], 16, 32);   // combine m-halves across lane pairs
    if (l < 16)
      pooled[(size_t)w * 128 + nt * 16 + l] = v * (1.f / 32.f) + projb[nt * 16 + l];
  }
}

// ---------- K3: cross-attention over neighbor windows + residual + LayerNorm ----------
__global__ __launch_bounds__(128) void k_cross(
    const float* __restrict__ pooled,
    const float* __restrict__ caiw, const float* __restrict__ caib,
    const float* __restrict__ caow, const float* __restrict__ caob,
    const float* __restrict__ lng, const float* __restrict__ lnb,
    float* __restrict__ seq)
{
  __shared__ float sm[4][2080];
  int wv = threadIdx.x >> 5, l = threadIdx.x & 31;
  int gid = blockIdx.x * 4 + wv;
  int b = gid >> 6, ci = gid & 63;
  float* S   = sm[wv];
  float* cen = S;          // 128
  float* ctx = S + 128;    // 4*128
  float* q   = S + 640;    // 128
  float* kk  = S + 768;    // 4*128
  float* vv  = S + 1280;   // 4*128
  float* scn = S + 1792;   // 16
  float* aw  = S + 1808;   // 16
  float* oo  = S + 1824;   // 128
  float* rr  = S + 1952;   // 128

  int left  = (ci - 2 < 0) ? 0 : ci - 2;
  int right = (ci + 3 > 64) ? 64 : ci + 3;
  int nb[4]; int nctx = 0;
  for (int i = left; i < right; ++i) if (i != ci) nb[nctx++] = i;

  for (int e = l; e < 128; e += 32) cen[e] = pooled[(size_t)(b * 64 + ci) * 128 + e];
  for (int c = 0; c < nctx; ++c)
    for (int e = l; e < 128; e += 32) ctx[c * 128 + e] = pooled[(size_t)(b * 64 + nb[c]) * 128 + e];

  const float* Wi = caiw + (size_t)ci * 384 * 128;
  const float* bi = caib + (size_t)ci * 384;
  for (int j = l; j < 128; j += 32) {
    float s = bi[j];
    const float* wr = Wi + (size_t)j * 128;
    for (int e = 0; e < 128; ++e) s += cen[e] * wr[e];
    q[j] = s;
  }
  for (int c = 0; c < nctx; ++c) {
    for (int j = l; j < 128; j += 32) {
      float sk = bi[128 + j], sv = bi[256 + j];
      const float* wk  = Wi + (size_t)(128 + j) * 128;
      const float* wvp = Wi + (size_t)(256 + j) * 128;
      for (int e = 0; e < 128; ++e) { float xe = ctx[c * 128 + e]; sk += xe * wk[e]; sv += xe * wvp[e]; }
      kk[c * 128 + j] = sk; vv[c * 128 + j] = sv;
    }
  }
  if (l < 16) {
    int h = l >> 2, c = l & 3;
    if (c < nctx) {
      float s = 0.f;
      for (int d = 0; d < 32; ++d) s += q[h * 32 + d] * kk[c * 128 + h * 32 + d];
      scn[l] = s * 0.17677669529663687f;
    }
  }
  if (l < 4) {
    int h = l;
    float mx = -3.0e38f;
    for (int c = 0; c < nctx; ++c) mx = fmaxf(mx, scn[h * 4 + c]);
    float ssum = 0.f;
    for (int c = 0; c < nctx; ++c) { float e = __expf(scn[h * 4 + c] - mx); aw[h * 4 + c] = e; ssum += e; }
    float inv = rcp_fast(ssum);
    for (int c = 0; c < nctx; ++c) aw[h * 4 + c] *= inv;
  }
  for (int j = l; j < 128; j += 32) {
    int h = j >> 5;
    float s = 0.f;
    for (int c = 0; c < nctx; ++c) s += aw[h * 4 + c] * vv[c * 128 + j];
    oo[j] = s;
  }
  const float* Wo = caow + (size_t)ci * 128 * 128;
  const float* bo = caob + (size_t)ci * 128;
  for (int j = l; j < 128; j += 32) {
    float s = bo[j];
    const float* wr = Wo + (size_t)j * 128;
    for (int e = 0; e < 128; ++e) s += oo[e] * wr[e];
    rr[j] = s + cen[j];
  }
  float part = 0.f;
  for (int j = l; j < 128; j += 32) part += rr[j];
#pragma unroll
  for (int off = 16; off >= 1; off >>= 1) part += __shfl_xor(part, off, 32);
  float mean = part * (1.f / 128.f);
  float vpart = 0.f;
  for (int j = l; j < 128; j += 32) { float d = rr[j] - mean; vpart += d * d; }
#pragma unroll
  for (int off = 16; off >= 1; off >>= 1) vpart += __shfl_xor(vpart, off, 32);
  float rstd = rsqrtf(vpart * (1.f / 128.f) + 1e-5f);
  for (int j = l; j < 128; j += 32)
    seq[(size_t)b * 8192 + ci * 128 + j] =
        (rr[j] - mean) * rstd * lng[ci * 128 + j] + lnb[ci * 128 + j];
}

// ---------- K4/K5: dense head ----------
__global__ __launch_bounds__(256) void k_fc1(const float* __restrict__ seq,
                                             const float* __restrict__ w,
                                             const float* __restrict__ b,
                                             float* __restrict__ h1)
{
  int id = blockIdx.x * 256 + threadIdx.x;        // 256*64
  if (id >= 256 * 64) return;
  int bb = id >> 6, j = id & 63;
  const float* sr = seq + (size_t)bb * 8192;
  const float* wr = w + (size_t)j * 8192;
  float s = b[j];
  for (int k = 0; k < 8192; ++k) s += sr[k] * wr[k];
  h1[id] = s > 0.f ? s : 0.f;
}

__global__ __launch_bounds__(256) void k_fc2(const float* __restrict__ h1,
                                             const float* __restrict__ w,
                                             const float* __restrict__ b,
                                             float* __restrict__ out)
{
  int id = blockIdx.x * 256 + threadIdx.x;        // 256*5
  if (id >= 256 * 5) return;
  int bb = id / 5, o = id % 5;
  const float* hr = h1 + (size_t)bb * 64;
  const float* wr = w + (size_t)o * 64;
  float s = b[o];
  for (int k = 0; k < 64; ++k) s += hr[k] * wr[k];
  out[id] = s;
}

// ---------- launch ----------
extern "C" void kernel_launch(void* const* d_in, const int* in_sizes, int n_in,
                              void* d_out, int out_size, void* d_ws, size_t ws_size,
                              hipStream_t stream) {
  (void)in_sizes; (void)n_in; (void)out_size; (void)ws_size;
  const float* x      = (const float*)d_in[0];
  const float* w_ih_f = (const float*)d_in[1];
  const float* w_hh_f = (const float*)d_in[2];
  const float* b_ih_f = (const float*)d_in[3];
  const float* b_hh_f = (const float*)d_in[4];
  const float* w_ih_b = (const float*)d_in[5];
  const float* w_hh_b = (const float*)d_in[6];
  const float* b_ih_b = (const float*)d_in[7];
  const float* b_hh_b = (const float*)d_in[8];
  const float* sa_in_w  = (const float*)d_in[9];
  const float* sa_in_b  = (const float*)d_in[10];
  const float* sa_out_w = (const float*)d_in[11];
  const float* sa_out_b = (const float*)d_in[12];
  const float* proj_w   = (const float*)d_in[13];
  const float* proj_b   = (const float*)d_in[14];
  const float* ca_in_w  = (const float*)d_in[15];
  const float* ca_in_b  = (const float*)d_in[16];
  const float* ca_out_w = (const float*)d_in[17];
  const float* ca_out_b = (const float*)d_in[18];
  const float* ln_g     = (const float*)d_in[19];
  const float* ln_b     = (const float*)d_in[20];
  const float* fd1_w    = (const float*)d_in[21];
  const float* fd1_b    = (const float*)d_in[22];
  const float* fd2_w    = (const float*)d_in[23];
  const float* fd2_b    = (const float*)d_in[24];

  char* ws = (char*)d_ws;
  size_t o_wih    = 0;
  size_t o_whh    = o_wih  + (size_t)2 * 256 * 32 * 2;      //  32768
  size_t o_bsum   = o_whh  + (size_t)2 * 256 * 64 * 2;      //  98304
  size_t o_sain   = o_bsum + (size_t)2 * 256 * 4;           // 100352
  size_t o_saout  = o_sain + (size_t)384 * 128 * 2;         // 198656
  size_t o_proj   = o_saout+ (size_t)128 * 128 * 2;         // 231424
  size_t o_lo     = o_proj + (size_t)128 * 128 * 2;         // 264192 (16B aligned)
  size_t o_pooled = o_lo   + (size_t)NWIN_TOT * 32 * 128 * 2;
  size_t o_seq    = o_pooled + (size_t)256 * 64 * 128 * 4;
  size_t o_h1     = o_seq  + (size_t)256 * 8192 * 4;

  _Float16* wih16  = (_Float16*)(ws + o_wih);
  _Float16* whh16  = (_Float16*)(ws + o_whh);
  float*    bsum   = (float*)   (ws + o_bsum);
  _Float16* sain16 = (_Float16*)(ws + o_sain);
  _Float16* saout16= (_Float16*)(ws + o_saout);
  _Float16* proj16 = (_Float16*)(ws + o_proj);
  _Float16* lo16   = (_Float16*)(ws + o_lo);
  float*    pooled = (float*)   (ws + o_pooled);
  float*    seq    = (float*)   (ws + o_seq);
  float*    h1     = (float*)   (ws + o_h1);

  k_prep<<<514, 256, 0, stream>>>(w_ih_f, w_hh_f, b_ih_f, b_hh_f,
                                  w_ih_b, w_hh_b, b_ih_b, b_hh_b,
                                  sa_in_w, sa_out_w, proj_w,
                                  wih16, whh16, bsum, sain16, saout16, proj16);
  k_bilstm<<<NWIN_TOT / 64, 128, 0, stream>>>(x, wih16, whh16, bsum, lo16);
  k_mha<<<NWIN_TOT, 32, 0, stream>>>(lo16, sain16, sa_in_b, saout16, sa_out_b,
                                     proj16, proj_b, pooled);
  k_cross<<<NWIN_TOT / 4, 128, 0, stream>>>(pooled, ca_in_w, ca_in_b,
                                            ca_out_w, ca_out_b, ln_g, ln_b, seq);
  k_fc1<<<64, 256, 0, stream>>>(seq, fd1_w, fd1_b, h1);
  k_fc2<<<5, 256, 0, stream>>>(h1, fd2_w, fd2_b, (float*)d_out);
}
